// WaveFieldAttentionV4_46024869544533
// MI455X (gfx1250) — matrix operand order repaired and checked by reading, not compiled, not run on hardware
//
#include <hip/hip_runtime.h>
#include <stdint.h>

// ---------------------------------------------------------------------------
// WaveFieldAttention for MI455X (gfx1250, wave32, WMMA + TDM).
// Sizes fixed by the reference: B=4, N=2048, D=1024, H=16, HD=64, G=4096.
// ---------------------------------------------------------------------------

#define BB 4
#define NN 2048
#define DD 1024
#define HH 16
#define HDD 64
#define GG 4096
#define MM (BB * NN)           // 8192 rows
#define CHUNK 512              // scan chunk length
#define NCHUNK (GG / CHUNK)    // 8
#define NLINES (BB * HH * HDD) // 4096 scan lines

#define KSLAB 64               // K elements staged per LDS slab
#define LDSPITCH 72            // 64 elems + 8 elems (16B) TDM pad -> 144B row pitch

typedef __bf16 bf16_t;
typedef __attribute__((ext_vector_type(16))) __bf16 v16bf;
typedef __attribute__((ext_vector_type(8)))  __bf16 v8bf;
typedef __attribute__((ext_vector_type(8)))  float  v8f;
typedef unsigned int u32x4 __attribute__((ext_vector_type(4)));
typedef int i32x4 __attribute__((ext_vector_type(4)));
typedef int i32x8 __attribute__((ext_vector_type(8)));

union Frag16 { v16bf v; v8bf h[2]; };

#if defined(__HIP_DEVICE_COMPILE__) && __has_builtin(__builtin_amdgcn_tensor_load_to_lds) && \
    __has_builtin(__builtin_amdgcn_s_wait_tensorcnt)
#define USE_TDM 1
#else
#define USE_TDM 0
#endif

#if USE_TDM
// ---------------------------------------------------------------------------
// Issue one TDM 2D tile load: global (tile_d1 rows x tile_d0 elems, row stride
// `stride_elems`, 2-byte elements) -> LDS at byte offset lds_off, with LDS
// padding of 4 DWORDs after every 32 DWORDs (row pitch 64 -> 72 elements).
// Tensor dims are declared equal to the tile (always in-bounds here).
// ---------------------------------------------------------------------------
__device__ inline void tdm_load_tile_2d(unsigned lds_off, const void* gptr,
                                        unsigned tile_d0, unsigned tile_d1,
                                        unsigned stride_elems) {
  unsigned long long ga = (unsigned long long)(uintptr_t)gptr;
  u32x4 g0;
  g0.x = 1u;                                            // count=1, user mode
  g0.y = lds_off;                                       // lds_addr (bytes)
  g0.z = (unsigned)(ga & 0xFFFFFFFFu);                  // global_addr[95:64]
  g0.w = (unsigned)((ga >> 32) & 0x1FFFFFFu) | (2u << 30); // addr hi | type=2
  i32x8 g1;
  g1[0] = (1 << 16)        // data_size = 1 -> 2 bytes
        | (1 << 20)        // pad_enable
        | (4 << 22)        // pad_interval: 32 DWORDs (128B) between pads
        | (3 << 25);       // pad_amount: 4 DWORDs (16B)
  g1[1] = (int)((tile_d0 & 0xFFFFu) << 16);                               // tensor_dim0 lo
  g1[2] = (int)(((tile_d0 >> 16) & 0xFFFFu) | ((tile_d1 & 0xFFFFu) << 16)); // dim0 hi | dim1 lo
  g1[3] = (int)(((tile_d1 >> 16) & 0xFFFFu) | ((tile_d0 & 0xFFFFu) << 16)); // dim1 hi | tile_dim0
  g1[4] = (int)(tile_d1 & 0xFFFFu);                                       // tile_dim1 | tile_dim2=0
  g1[5] = (int)stride_elems;                                              // tensor_dim0_stride lo32
  g1[6] = 0;                                                              // stride hi | dim1_stride lo
  g1[7] = 0;
  i32x4 g2; g2[0] = 1; g2[1] = 0; g2[2] = 0; g2[3] = 0;  // tensor_dim2=1, rest 0
  i32x4 g3; g3[0] = 0; g3[1] = 0; g3[2] = 0; g3[3] = 0;
#if __clang_major__ >= 23
  i32x8 gz = {0, 0, 0, 0, 0, 0, 0, 0};
  __builtin_amdgcn_tensor_load_to_lds(g0, g1, g2, g3, gz, 0);
#else
  __builtin_amdgcn_tensor_load_to_lds(g0, g1, g2, g3, 0);
#endif
}
#endif

// ---------------------------------------------------------------------------
// float -> bf16 flat convert
// ---------------------------------------------------------------------------
__global__ __launch_bounds__(256) void cvt_bf16_kernel(const float* __restrict__ in,
                                                       bf16_t* __restrict__ out,
                                                       size_t n) {
  size_t i = (size_t)blockIdx.x * blockDim.x + threadIdx.x;
  if (i < n) out[i] = (bf16_t)in[i];
}

// W (K x Ncols slice of row-major [K x ldw], starting at colOff)
// -> WT bf16, row-major (Ncols x K): transposed so B-fragments are contiguous.
__global__ __launch_bounds__(256) void cvt_wT_kernel(const float* __restrict__ W,
                                                     bf16_t* __restrict__ WT,
                                                     int K, int Ncols, int ldw, int colOff) {
  size_t i = (size_t)blockIdx.x * blockDim.x + threadIdx.x;
  if (i >= (size_t)K * Ncols) return;
  int n = (int)(i % Ncols);
  int k = (int)(i / Ncols);
  WT[(size_t)n * K + k] = (bf16_t)W[(size_t)k * ldw + colOff + n];
}

// ---------------------------------------------------------------------------
// Per-head IIR constants + coupling softmax.
// consts: [h*8 + {0:r_re 1:r_im 2:c_re 3:c_im 4:inv_norm 5:rL_re 6:rL_im}],
// coupling (softmax rows) at consts+128 (256 floats).
// ---------------------------------------------------------------------------
__global__ __launch_bounds__(64) void setup_kernel(const float* __restrict__ wfreq,
                                                   const float* __restrict__ wdamp,
                                                   const float* __restrict__ wphase,
                                                   const float* __restrict__ fc,
                                                   float* __restrict__ consts) {
  int t = threadIdx.x;
  if (t < HH) {
    float alpha = log1pf(expf(wdamp[t])) + 0.05f;   // softplus + FLOOR
    float omega = fabsf(wfreq[t]);
    float phi   = wphase[t];
    float e  = expf(-alpha);
    float rr = e * cosf(omega);
    float ri = e * sinf(omega);
    float norm = 0.f;
    for (int g = 0; g < GG; ++g)
      norm += fabsf(expf(-alpha * (float)g) * cosf(omega * (float)g + phi));
    norm = fmaxf(norm, 1e-8f);
    float ar = rr, ai = ri;                         // rL = r^CHUNK (2^9)
    for (int s = 0; s < 9; ++s) { float nr = ar*ar - ai*ai; ai = 2.f*ar*ai; ar = nr; }
    consts[t*8+0] = rr;  consts[t*8+1] = ri;
    consts[t*8+2] = cosf(phi); consts[t*8+3] = sinf(phi);
    consts[t*8+4] = 1.f / norm;
    consts[t*8+5] = ar;  consts[t*8+6] = ai;
    float m = -1e30f;                               // softmax row t of coupling
    for (int j = 0; j < HH; ++j) m = fmaxf(m, fc[t*HH+j]);
    float s = 0.f, tmp[HH];
    for (int j = 0; j < HH; ++j) { tmp[j] = expf(fc[t*HH+j] - m); s += tmp[j]; }
    for (int j = 0; j < HH; ++j) consts[128 + t*HH + j] = tmp[j] / s;
  }
}

// ---------------------------------------------------------------------------
// Zero fill (float4)
// ---------------------------------------------------------------------------
__global__ __launch_bounds__(256) void zero_kernel(float4* __restrict__ p, size_t n4) {
  size_t i = (size_t)blockIdx.x * blockDim.x + threadIdx.x;
  if (i < n4) p[i] = make_float4(0.f, 0.f, 0.f, 0.f);
}

// ---------------------------------------------------------------------------
// bf16 WMMA GEMM, TDM double-buffered through LDS.
// C[M x N] (fp32) = A[M x K] * BT[N x K]^T  (+ bias epilogue)
// Block: 256 thr = 8 waves, tile 128x128. Wave: 32(M) x 64(N) = 2x4 WMMA tiles.
// Per slab (K=64): wave0 issues 2 tensor_load_to_lds (A,B) one slab ahead,
// waits TENSORcnt, workgroup barrier, fragments read from LDS (ds_load_b128).
// EPI: 0 = C = acc + bias ; 1 = C = sigmoid(acc + bias)
// ---------------------------------------------------------------------------
template <int EPI>
__global__ __launch_bounds__(256) void gemm_bf16_kernel(const bf16_t* __restrict__ A,
                                                        const bf16_t* __restrict__ BT,
                                                        float* __restrict__ C,
                                                        const float* __restrict__ bias,
                                                        int M, int N, int K) {
  __shared__ bf16_t sA[2][128][LDSPITCH];
  __shared__ bf16_t sB[2][128][LDSPITCH];

  const int lane = threadIdx.x & 31;
  const int wid  = threadIdx.x >> 5;
  const int wm   = wid & 3;            // 4 waves down M
  const int wn   = wid >> 2;           // 2 waves across N
  const int blockM = blockIdx.x * 128;
  const int blockN = blockIdx.y * 128;
  const int r    = lane & 15;
  const int half = lane >> 4;

  v8f acc[2][4];
  const v8f vzero = {0.f,0.f,0.f,0.f,0.f,0.f,0.f,0.f};
#pragma unroll
  for (int i = 0; i < 2; ++i)
#pragma unroll
    for (int j = 0; j < 4; ++j) acc[i][j] = vzero;

  const int nslab = K / KSLAB;

#if USE_TDM
  if (wid == 0) {
    tdm_load_tile_2d((unsigned)(uintptr_t)&sA[0][0][0],
                     A + (size_t)blockM * K, KSLAB, 128, (unsigned)K);
    tdm_load_tile_2d((unsigned)(uintptr_t)&sB[0][0][0],
                     BT + (size_t)blockN * K, KSLAB, 128, (unsigned)K);
  }
#endif

  for (int s = 0; s < nslab; ++s) {
    const int buf = s & 1;
#if USE_TDM
    if (wid == 0) {
      if (s + 1 < nslab) {
        const bf16_t* gA = A  + (size_t)blockM * K + (size_t)(s + 1) * KSLAB;
        const bf16_t* gB = BT + (size_t)blockN * K + (size_t)(s + 1) * KSLAB;
        tdm_load_tile_2d((unsigned)(uintptr_t)&sA[buf ^ 1][0][0], gA, KSLAB, 128, (unsigned)K);
        tdm_load_tile_2d((unsigned)(uintptr_t)&sB[buf ^ 1][0][0], gB, KSLAB, 128, (unsigned)K);
        __builtin_amdgcn_s_wait_tensorcnt(2);   // slab s (2 oldest) complete
      } else {
        __builtin_amdgcn_s_wait_tensorcnt(0);   // final slab complete
      }
    }
#else
    {
      const bf16_t* gA = A  + (size_t)blockM * K + (size_t)s * KSLAB;
      const bf16_t* gB = BT + (size_t)blockN * K + (size_t)s * KSLAB;
      for (int idx = threadIdx.x; idx < 128 * (KSLAB / 8); idx += 256) {
        int row = idx / (KSLAB / 8);
        int c8  = (idx % (KSLAB / 8)) * 8;
        *(v8bf*)&sA[buf][row][c8] = *(const v8bf*)(gA + (size_t)row * K + c8);
        *(v8bf*)&sB[buf][row][c8] = *(const v8bf*)(gB + (size_t)row * K + c8);
      }
    }
#endif
    __syncthreads();

#pragma unroll
    for (int ks = 0; ks < KSLAB; ks += 32) {
      Frag16 a[2];
#pragma unroll
      for (int i = 0; i < 2; ++i) {
        // A 16x32 bf16 frag: lanes 0-15 row=r hold K {0..7,16..23}, lanes 16-31 {8..15,24..31}
        const bf16_t* p = &sA[buf][wm * 32 + i * 16 + r][ks + half * 8];
        a[i].h[0] = *(const v8bf*)(p);
        a[i].h[1] = *(const v8bf*)(p + 16);
      }
      Frag16 b[4];
#pragma unroll
      for (int j = 0; j < 4; ++j) {
        // B 32x16 bf16 frag: lanes 0-15 col=r hold K {0..7,16..23}, lanes 16-31 {8..15,24..31}
        const bf16_t* p = &sB[buf][wn * 64 + j * 16 + r][ks + half * 8];
        b[j].h[0] = *(const v8bf*)(p);
        b[j].h[1] = *(const v8bf*)(p + 16);
      }
#pragma unroll
      for (int i = 0; i < 2; ++i)
#pragma unroll
        for (int j = 0; j < 4; ++j)
          acc[i][j] = __builtin_amdgcn_wmma_f32_16x16x32_bf16(
              false, a[i].v, false, b[j].v, (short)0, acc[i][j], false, false);
    }
    __syncthreads();
  }

  // C/D layout: VGPR v -> row v (lanes 0-15) / row v+8 (lanes 16-31), col = lane&15.
#pragma unroll
  for (int i = 0; i < 2; ++i) {
#pragma unroll
    for (int j = 0; j < 4; ++j) {
      int col = blockN + wn * 64 + j * 16 + r;
      float bv = bias ? bias[col] : 0.f;
#pragma unroll
      for (int v = 0; v < 8; ++v) {
        int row = blockM + wm * 32 + i * 16 + half * 8 + v;
        float val = acc[i][j][v] + bv;
        if (EPI == 1) val = 1.f / (1.f + __expf(-val));
        C[(size_t)row * N + col] = val;
      }
    }
  }
}

// ---------------------------------------------------------------------------
// Scatter: deposit = v * ||k|| into field at lo/lo+1 with linear weights.
// Block = one (b,n); lo gaps >= 2 (stride ~2.0005) so plain RMW is race-free.
// ---------------------------------------------------------------------------
__global__ __launch_bounds__(256) void scatter_kernel(const float* __restrict__ kv,
                                                      float* __restrict__ field) {
  int bn = blockIdx.x;
  int b = bn / NN, n = bn % NN;
  int t = threadIdx.x;
  int h = t >> 4;
  int dbase = (t & 15) * 4;

  const float* krow = kv + (size_t)bn * (2 * DD);
  const float* vrow = krow + DD;

  float ss = 0.f;
#pragma unroll
  for (int q = 0; q < 4; ++q) {
    float kk = krow[h * HDD + dbase + q];
    ss += kk * kk;
  }
#pragma unroll
  for (int m = 8; m >= 1; m >>= 1) ss += __shfl_xor(ss, m, 32);
  float kmag = sqrtf(ss);

  float stride = (float)(GG - 1) / (float)(NN - 1);
  float pos = fminf((float)n * stride, (float)(GG - 2));
  int lo = (int)pos;
  float frac = pos - (float)lo;
  float w0 = 1.f - frac, w1 = frac;

  float* f0 = field + (((size_t)(b * HH + h)) * GG + lo) * HDD;
#pragma unroll
  for (int q = 0; q < 4; ++q) {
    int d = dbase + q;
    float dep = vrow[h * HDD + d] * kmag;
    f0[d]       += w0 * dep;
    f0[HDD + d] += w1 * dep;
  }
}

// ---------------------------------------------------------------------------
// Causal IIR scan along G (exact replacement of the FFT convolution):
//   s[g] = r*s[g-1] + x[g] ;  y[g] = Re(e^{i phi} * s[g]) / ||ker||_1
// Phase 1: chunk-local scan, in-place, stores chunk end states.
// Phase 2: prefix-combine chunk states, add Re(c * r^{t+1} * S_in) correction.
// ---------------------------------------------------------------------------
__global__ __launch_bounds__(256) void scan1_kernel(float* __restrict__ field,
                                                    const float* __restrict__ consts,
                                                    float* __restrict__ states) {
  int tid = blockIdx.x * blockDim.x + threadIdx.x;
  if (tid >= NLINES * NCHUNK) return;
  int chunk = tid / NLINES;
  int line  = tid % NLINES;
  int d  = line & (HDD - 1);
  int bh = line / HDD;
  int h  = bh & (HH - 1);

  float rr = consts[h*8+0], ri = consts[h*8+1];
  float cr = consts[h*8+2], ci = consts[h*8+3], inv = consts[h*8+4];

  float* base = field + ((size_t)bh * GG) * HDD + d;
  int g0 = chunk * CHUNK;
  float sr = 0.f, si = 0.f;
  for (int t = 0; t < CHUNK; ++t) {
    size_t idx = (size_t)(g0 + t) * HDD;
    float x = base[idx];
    float nsr = rr*sr - ri*si + x;
    si = rr*si + ri*sr;
    sr = nsr;
    base[idx] = (cr*sr - ci*si) * inv;
  }
  states[(size_t)tid * 2]     = sr;
  states[(size_t)tid * 2 + 1] = si;
}

__global__ __launch_bounds__(256) void scan2_kernel(float* __restrict__ field,
                                                    const float* __restrict__ consts,
                                                    const float* __restrict__ states) {
  int tid = blockIdx.x * blockDim.x + threadIdx.x;
  if (tid >= NLINES * NCHUNK) return;
  int chunk = tid / NLINES;
  if (chunk == 0) return;
  int line  = tid % NLINES;
  int d  = line & (HDD - 1);
  int bh = line / HDD;
  int h  = bh & (HH - 1);

  float rr = consts[h*8+0], ri = consts[h*8+1];
  float cr = consts[h*8+2], ci = consts[h*8+3], inv = consts[h*8+4];
  float rLr = consts[h*8+5], rLi = consts[h*8+6];

  float Sr = 0.f, Si = 0.f;                       // state entering this chunk
  for (int j = 0; j < chunk; ++j) {
    float er = states[((size_t)j * NLINES + line) * 2];
    float ei = states[((size_t)j * NLINES + line) * 2 + 1];
    float nr = rLr*Sr - rLi*Si + er;
    Si = rLr*Si + rLi*Sr + ei;
    Sr = nr;
  }

  float* base = field + ((size_t)bh * GG) * HDD + d;
  int g0 = chunk * CHUNK;
  float pr = Sr, pi = Si;                         // p = r^{t+1} * S
  for (int t = 0; t < CHUNK; ++t) {
    float npr = rr*pr - ri*pi;
    pi = rr*pi + ri*pr;
    pr = npr;
    size_t idx = (size_t)(g0 + t) * HDD;
    base[idx] += (cr*pr - ci*pi) * inv;
  }
}

// ---------------------------------------------------------------------------
// Gather at lo/lo+1, mix heads with softmax coupling, multiply by gate,
// emit bf16 A-matrix for the final GEMM.  Block = one (b,n).
// ---------------------------------------------------------------------------
__global__ __launch_bounds__(256) void gather_kernel(const float* __restrict__ field,
                                                     const float* __restrict__ gate,
                                                     const float* __restrict__ consts,
                                                     bf16_t* __restrict__ ga) {
  int bn = blockIdx.x;
  int b = bn / NN, n = bn % NN;
  float stride = (float)(GG - 1) / (float)(NN - 1);
  float pos = fminf((float)n * stride, (float)(GG - 2));
  int lo = (int)pos;
  float frac = pos - (float)lo;
  float w0 = 1.f - frac, w1 = frac;

  __shared__ float f0[HH][HDD];
  __shared__ float f1[HH][HDD];
  __shared__ float cpl[HH * HH];

  int t = threadIdx.x;
  cpl[t] = consts[128 + t];
  for (int idx = t; idx < HH * HDD; idx += 256) {
    int j = idx >> 6, d = idx & (HDD - 1);
    const float* p = field + (((size_t)(b * HH + j)) * GG + lo) * HDD + d;
    f0[j][d] = p[0];
    f1[j][d] = p[HDD];
  }
  __syncthreads();

  for (int idx = t; idx < HH * HDD; idx += 256) {
    int i = idx >> 6, d = idx & (HDD - 1);
    float acc = 0.f;
#pragma unroll
    for (int j = 0; j < HH; ++j)
      acc += cpl[i * HH + j] * (w0 * f0[j][d] + w1 * f1[j][d]);
    float g = gate[(size_t)bn * DD + idx];
    ga[(size_t)bn * DD + idx] = (bf16_t)(acc * g);
  }
}

// ---------------------------------------------------------------------------
// Launch
// ---------------------------------------------------------------------------
extern "C" void kernel_launch(void* const* d_in, const int* in_sizes, int n_in,
                              void* d_out, int out_size, void* d_ws, size_t ws_size,
                              hipStream_t stream) {
  const float* x      = (const float*)d_in[0];
  const float* Wqkv   = (const float*)d_in[1];
  const float* bqkv   = (const float*)d_in[2];
  const float* Wout   = (const float*)d_in[3];
  const float* bout   = (const float*)d_in[4];
  const float* Wgate  = (const float*)d_in[5];
  const float* bgate  = (const float*)d_in[6];
  const float* wfreq  = (const float*)d_in[7];
  const float* wdamp  = (const float*)d_in[8];
  const float* wphase = (const float*)d_in[9];
  const float* fc     = (const float*)d_in[10];
  float* out = (float*)d_out;

  char* ws = (char*)d_ws;
  size_t off = 0;
  auto take = [&](size_t bytes) -> void* {
    void* p = ws + off;
    off = (off + bytes + 255) & ~(size_t)255;
    return p;
  };

  bf16_t* xb     = (bf16_t*)take((size_t)MM * DD * 2);            // 16 MB
  bf16_t* wkvT   = (bf16_t*)take((size_t)(2 * DD) * DD * 2);      //  4 MB
  bf16_t* wgateT = (bf16_t*)take((size_t)DD * DD * 2);            //  2 MB
  bf16_t* woutT  = (bf16_t*)take((size_t)DD * DD * 2);            //  2 MB
  float*  kvbuf  = (float*)take((size_t)MM * 2 * DD * 4);         // 64 MB
  float*  gatef  = (float*)take((size_t)MM * DD * 4);             // 32 MB
  float*  field  = (float*)take((size_t)BB * HH * GG * HDD * 4);  // 64 MB
  bf16_t* ga     = (bf16_t*)take((size_t)MM * DD * 2);            // 16 MB
  float*  states = (float*)take((size_t)NLINES * NCHUNK * 2 * 4); // 256 KB
  float*  consts = (float*)take(4096);

  // 1) precision convert / transpose (q projection skipped: unused by reference)
  {
    size_t n = (size_t)MM * DD;
    cvt_bf16_kernel<<<dim3((unsigned)((n + 255) / 256)), dim3(256), 0, stream>>>(x, xb, n);
  }
  cvt_wT_kernel<<<dim3((DD * 2 * DD + 255) / 256), dim3(256), 0, stream>>>(
      Wqkv, wkvT, DD, 2 * DD, 3 * DD, DD);
  cvt_wT_kernel<<<dim3((DD * DD + 255) / 256), dim3(256), 0, stream>>>(
      Wgate, wgateT, DD, DD, DD, 0);
  cvt_wT_kernel<<<dim3((DD * DD + 255) / 256), dim3(256), 0, stream>>>(
      Wout, woutT, DD, DD, DD, 0);

  // 2) per-head IIR constants + coupling softmax
  setup_kernel<<<dim3(1), dim3(64), 0, stream>>>(wfreq, wdamp, wphase, fc, consts);

  // 3) zero the field
  {
    size_t n4 = (size_t)BB * HH * GG * HDD / 4;
    zero_kernel<<<dim3((unsigned)((n4 + 255) / 256)), dim3(256), 0, stream>>>((float4*)field, n4);
  }

  // 4) kv = x @ Wqkv[:, D:3D] + bqkv[D:]   (M=8192, N=2048, K=1024)
  gemm_bf16_kernel<0><<<dim3(MM / 128, (2 * DD) / 128), dim3(256), 0, stream>>>(
      xb, wkvT, kvbuf, bqkv + DD, MM, 2 * DD, DD);

  // 5) gate = sigmoid(x @ Wgate + bgate)
  gemm_bf16_kernel<1><<<dim3(MM / 128, DD / 128), dim3(256), 0, stream>>>(
      xb, wgateT, gatef, bgate, MM, DD, DD);

  // 6) scatter deposits into field
  scatter_kernel<<<dim3(BB * NN), dim3(256), 0, stream>>>(kvbuf, field);

  // 7) causal IIR scan (== FFT conv with damped-cosine kernel), chunked
  scan1_kernel<<<dim3(NLINES * NCHUNK / 256), dim3(256), 0, stream>>>(field, consts, states);
  scan2_kernel<<<dim3(NLINES * NCHUNK / 256), dim3(256), 0, stream>>>(field, consts, states);

  // 8) gather + head coupling + gate -> bf16 GEMM input
  gather_kernel<<<dim3(BB * NN), dim3(256), 0, stream>>>(field, gatef, consts, ga);

  // 9) out = (gathered*gate) @ Wout + bout
  gemm_bf16_kernel<0><<<dim3(MM / 128, DD / 128), dim3(256), 0, stream>>>(
      ga, woutT, out, bout, MM, DD, DD);
}
